// CausalSelfAttention_59571196396116
// MI455X (gfx1250) — matrix operand order repaired and checked by reading, not compiled, not run on hardware
//
#include <hip/hip_runtime.h>
#include <hip/hip_bf16.h>
#include <math.h>

// ---------------------------------------------------------------------------
// CausalSelfAttention with T5 relative bias, MI455X (gfx1250, wave32, WMMA)
// fp32 in/out, bf16 tensor-core compute with fp32 accumulation.
// Round 2: software-pipelined GEMMs (double-buffered fragments, 32x32 wave
// tiles) and pipelined K/V fragment loads in the flash-attention kernel.
// ---------------------------------------------------------------------------

typedef __bf16 bf16_t;
typedef __bf16 v16bf __attribute__((ext_vector_type(16)));
typedef __bf16 v8bf  __attribute__((ext_vector_type(8)));
typedef float  v8f   __attribute__((ext_vector_type(8)));

#define WMMA_BF16(a, b, c) \
  __builtin_amdgcn_wmma_f32_16x16x32_bf16(false, (a), false, (b), (short)0, (c), false, false)

#define NEG_INF (-1.0e9f)

union Frag { v16bf v; v8bf h[2]; };

// A-fragment (16x32 bf16, MxK), A points at tile origin, row-major, stride lda.
// Lane L<16: M=L, K={0..7,16..23}; Lane L>=16: M=L-16, K={8..15,24..31}.
__device__ __forceinline__ v16bf load_a_frag(const bf16_t* A, int lda, int lane) {
  Frag f;
  const bf16_t* p = A + (lane & 15) * lda + ((lane & 16) ? 8 : 0);
  f.h[0] = *(const v8bf*)(p);
  f.h[1] = *(const v8bf*)(p + 16);
  return f.v;
}

// B-fragment (32x16 bf16, KxN) loaded from column-major B, i.e. Bt[N][K] row-major.
// Lane L<16: N=L, K={0..7,16..23}; Lane L>=16: N=L-16, K={8..15,24..31}.
__device__ __forceinline__ v16bf load_b_frag(const bf16_t* Bt, int ldb, int lane) {
  Frag f;
  const bf16_t* p = Bt + (lane & 15) * ldb + ((lane & 16) ? 8 : 0);
  f.h[0] = *(const v8bf*)(p);
  f.h[1] = *(const v8bf*)(p + 16);
  return f.v;
}

// ------------------------- conversion / setup kernels ----------------------

__global__ void k_cvt(const float* __restrict__ in, bf16_t* __restrict__ out, int n) {
  int i = blockIdx.x * blockDim.x + threadIdx.x;
  if (i < n) out[i] = (bf16_t)in[i];
}

// in[R][Cc] fp32  ->  out[Cc][R] bf16; index over OUTPUT for coalesced stores.
__global__ void k_cvt_T(const float* __restrict__ in, bf16_t* __restrict__ out,
                        int R, int Cc) {
  int i = blockIdx.x * blockDim.x + threadIdx.x;     // i = c*R + r
  if (i < R * Cc) {
    int c = i / R, r = i % R;
    out[i] = (bf16_t)in[r * Cc + c];
  }
}

// T5 causal bucketing, precomputed per (head, distance): biasd[16][2048]
__global__ void k_bias(const float* __restrict__ rel, float* __restrict__ biasd) {
  int dist = blockIdx.x * blockDim.x + threadIdx.x;
  if (dist >= 2048) return;
  int bucket;
  if (dist < 16) {
    bucket = dist;
  } else {
    float l = logf((float)dist * (1.0f / 16.0f)) * (16.0f / logf(128.0f));
    bucket = 16 + (int)l;               // trunc toward zero, matches astype(int32)
    if (bucket > 31) bucket = 31;
  }
  for (int h = 0; h < 16; ++h) biasd[h * 2048 + dist] = rel[bucket * 16 + h];
}

// ------------------------------ QKV GEMM -----------------------------------
// Xb[4096][1024] bf16  @  (WT[3072][1024] bf16)^T  -> scatter q/k/v.
// One wave computes 32(M) x 32(N); K-step 32; double-buffered fragments.
__global__ void k_gemm_qkv(const bf16_t* __restrict__ Xb, const bf16_t* __restrict__ WT,
                           bf16_t* __restrict__ Qb, bf16_t* __restrict__ Kb,
                           bf16_t* __restrict__ Vt) {
  const int lane = threadIdx.x & 31;
  const int wid  = (blockIdx.x * blockDim.x + threadIdx.x) >> 5;
  const int NT = 3072 / 32;
  const int tm = wid / NT, tn = wid % NT;
  if (tm >= 4096 / 32) return;
  const int m0 = tm * 32, n0 = tn * 32;

  const bf16_t* Ar0 = Xb + (size_t)m0 * 1024;
  const bf16_t* Ar1 = Xb + (size_t)(m0 + 16) * 1024;
  const bf16_t* Br0 = WT + (size_t)n0 * 1024;
  const bf16_t* Br1 = WT + (size_t)(n0 + 16) * 1024;

  v8f acc[2][2] = {};
  v16bf a0 = load_a_frag(Ar0, 1024, lane);
  v16bf a1 = load_a_frag(Ar1, 1024, lane);
  v16bf b0 = load_b_frag(Br0, 1024, lane);
  v16bf b1 = load_b_frag(Br1, 1024, lane);

  for (int k0 = 32; k0 < 1024; k0 += 32) {
    v16bf na0 = load_a_frag(Ar0 + k0, 1024, lane);
    v16bf na1 = load_a_frag(Ar1 + k0, 1024, lane);
    v16bf nb0 = load_b_frag(Br0 + k0, 1024, lane);
    v16bf nb1 = load_b_frag(Br1 + k0, 1024, lane);
    acc[0][0] = WMMA_BF16(a0, b0, acc[0][0]);
    acc[0][1] = WMMA_BF16(a0, b1, acc[0][1]);
    acc[1][0] = WMMA_BF16(a1, b0, acc[1][0]);
    acc[1][1] = WMMA_BF16(a1, b1, acc[1][1]);
    a0 = na0; a1 = na1; b0 = nb0; b1 = nb1;
  }
  acc[0][0] = WMMA_BF16(a0, b0, acc[0][0]);
  acc[0][1] = WMMA_BF16(a0, b1, acc[0][1]);
  acc[1][0] = WMMA_BF16(a1, b0, acc[1][0]);
  acc[1][1] = WMMA_BF16(a1, b1, acc[1][1]);

  const int roff = (lane & 16) ? 8 : 0;
#pragma unroll
  for (int ni = 0; ni < 2; ++ni) {
    int n = n0 + 16 * ni + (lane & 15);
    int sec = n >> 10;          // 0=q, 1=k, 2=v
    int c = n & 1023;
    int h = c >> 6, dd = c & 63;
#pragma unroll
    for (int mi = 0; mi < 2; ++mi) {
      int mbase = m0 + 16 * mi + roff;
#pragma unroll
      for (int i = 0; i < 8; ++i) {
        int m = mbase + i;
        int b = m >> 11, tt = m & 2047;
        int bh = b * 16 + h;
        float val = acc[mi][ni][i];
        if (sec == 0)      Qb[(bh * 2048 + tt) * 64 + dd] = (bf16_t)(val * 0.125f); // 1/sqrt(d)
        else if (sec == 1) Kb[(bh * 2048 + tt) * 64 + dd] = (bf16_t)val;
        else               Vt[(bh * 64 + dd) * 2048 + tt] = (bf16_t)val;            // V^T
      }
    }
  }
}

// ------------------------------ Flash attention ----------------------------
// One wave per (b, h, 16-query tile). Online softmax over 32-key blocks.
// K fragments double-buffered across blocks; V fragments loaded before the
// softmax VALU chain so their latency hides under it.
__global__ void k_attn(const bf16_t* __restrict__ Qb, const bf16_t* __restrict__ Kb,
                       const bf16_t* __restrict__ Vt, const float* __restrict__ biasd,
                       bf16_t* __restrict__ Y) {
  __shared__ bf16_t plds[8][16][40];   // per-wave 16x32 P tile, 80B row stride (16B aligned)

  const int lane  = threadIdx.x & 31;
  const int wslot = threadIdx.x >> 5;
  const int wid   = (blockIdx.x * blockDim.x + threadIdx.x) >> 5;  // 0..4095
  const int qt = wid & 127;
  const int bh = wid >> 7;             // b*16 + h
  const int h  = bh & 15;
  const int qbase = qt * 16;

  const bf16_t* Qp = Qb + (size_t)bh * 2048 * 64;
  const bf16_t* Kp = Kb + (size_t)bh * 2048 * 64;
  const bf16_t* Vp = Vt + (size_t)bh * 64 * 2048;

  const v16bf aq0 = load_a_frag(Qp + qbase * 64 + 0,  64, lane);
  const v16bf aq1 = load_a_frag(Qp + qbase * 64 + 32, 64, lane);

  v8f acc[4] = {};
  float rmax[8], rsum[8];
#pragma unroll
  for (int i = 0; i < 8; ++i) { rmax[i] = -3.0e38f; rsum[i] = 0.0f; }

  const int roff = (lane & 16) ? 8 : 0;
  const int ncol = lane & 15;
  const int nkb  = (qbase + 16 + 31) >> 5;     // causal: only keys <= qbase+15
  const int kend = nkb * 32;

  // preload K fragments for the first key block
  v16bf bk0 = load_b_frag(Kp + 0 * 64 + 0,  64, lane);
  v16bf bk1 = load_b_frag(Kp + 0 * 64 + 32, 64, lane);
  v16bf bk2 = load_b_frag(Kp + 16 * 64 + 0,  64, lane);
  v16bf bk3 = load_b_frag(Kp + 16 * 64 + 32, 64, lane);

  for (int kb = 0; kb < kend; kb += 32) {
    // ---- V fragments for this block (independent of scores; issue early) ----
    v16bf fv[4];
#pragma unroll
    for (int t = 0; t < 4; ++t) {
      Frag f;
      const bf16_t* p = Vp + (16 * t + (lane & 15)) * 2048 + kb + ((lane & 16) ? 8 : 0);
      f.h[0] = *(const v8bf*)(p);
      f.h[1] = *(const v8bf*)(p + 16);
      fv[t] = f.v;
    }
    // ---- scores: two 16x16 tiles, contraction over d=64 ----
    v8f s[2] = {};
    s[0] = WMMA_BF16(aq0, bk0, s[0]);
    s[0] = WMMA_BF16(aq1, bk1, s[0]);
    s[1] = WMMA_BF16(aq0, bk2, s[1]);
    s[1] = WMMA_BF16(aq1, bk3, s[1]);
    // ---- prefetch next block's K fragments (hidden under softmax) ----
    {
      int kbn = (kb + 32 < kend) ? kb + 32 : kb;
      bk0 = load_b_frag(Kp + (kbn) * 64 + 0,  64, lane);
      bk1 = load_b_frag(Kp + (kbn) * 64 + 32, 64, lane);
      bk2 = load_b_frag(Kp + (kbn + 16) * 64 + 0,  64, lane);
      bk3 = load_b_frag(Kp + (kbn + 16) * 64 + 32, 64, lane);
    }
    // ---- T5 bias + causal mask (scores already scaled via Q) ----
#pragma unroll
    for (int j = 0; j < 2; ++j) {
      int kk = kb + 16 * j + ncol;
#pragma unroll
      for (int i = 0; i < 8; ++i) {
        int q = qbase + roff + i;
        s[j][i] += (kk <= q) ? biasd[h * 2048 + (q - kk)] : NEG_INF;
      }
    }
    // ---- online softmax (row stats live in 16-lane half-groups) ----
#pragma unroll
    for (int i = 0; i < 8; ++i) {
      float tmax = fmaxf(s[0][i], s[1][i]);
#pragma unroll
      for (int m = 8; m >= 1; m >>= 1) tmax = fmaxf(tmax, __shfl_xor(tmax, m, 32));
      float mnew = fmaxf(rmax[i], tmax);
      float corr = __expf(rmax[i] - mnew);
      float p0 = __expf(s[0][i] - mnew);
      float p1 = __expf(s[1][i] - mnew);
      s[0][i] = p0; s[1][i] = p1;
      float ps = p0 + p1;
#pragma unroll
      for (int m = 8; m >= 1; m >>= 1) ps += __shfl_xor(ps, m, 32);
      rsum[i] = rsum[i] * corr + ps;
      rmax[i] = mnew;
#pragma unroll
      for (int t = 0; t < 4; ++t) acc[t][i] *= corr;
    }
    // ---- re-layout P: C-format -> LDS -> A-fragment (per-wave, no barrier) ----
#pragma unroll
    for (int j = 0; j < 2; ++j)
#pragma unroll
      for (int i = 0; i < 8; ++i)
        plds[wslot][roff + i][16 * j + ncol] = (bf16_t)s[j][i];

    Frag pa;
    {
      const bf16_t* p = &plds[wslot][lane & 15][(lane & 16) ? 8 : 0];
      pa.h[0] = *(const v8bf*)(p);
      pa.h[1] = *(const v8bf*)(p + 16);
    }
    // ---- PV: P(16x32) @ V(32x64); V loads long since issued ----
#pragma unroll
    for (int t = 0; t < 4; ++t) acc[t] = WMMA_BF16(pa.v, fv[t], acc[t]);
  }

  // ---- normalize and store Y[B*T][C] bf16 (layout for proj GEMM) ----
  const int b = bh >> 4;
#pragma unroll
  for (int t = 0; t < 4; ++t) {
    int dd = 16 * t + (lane & 15);
#pragma unroll
    for (int i = 0; i < 8; ++i) {
      int q = qbase + roff + i;
      Y[((size_t)b * 2048 + q) * 1024 + h * 64 + dd] = (bf16_t)(acc[t][i] / rsum[i]);
    }
  }
}

// ------------------------------ output proj --------------------------------
// Yb[4096][1024] @ (WT[1024][1024])^T -> Out fp32. 32x32 wave tile, pipelined.
__global__ void k_gemm_proj(const bf16_t* __restrict__ Yb, const bf16_t* __restrict__ WT,
                            float* __restrict__ Out) {
  const int lane = threadIdx.x & 31;
  const int wid  = (blockIdx.x * blockDim.x + threadIdx.x) >> 5;
  const int NT = 1024 / 32;
  const int tm = wid / NT, tn = wid % NT;
  if (tm >= 4096 / 32) return;
  const int m0 = tm * 32, n0 = tn * 32;

  const bf16_t* Ar0 = Yb + (size_t)m0 * 1024;
  const bf16_t* Ar1 = Yb + (size_t)(m0 + 16) * 1024;
  const bf16_t* Br0 = WT + (size_t)n0 * 1024;
  const bf16_t* Br1 = WT + (size_t)(n0 + 16) * 1024;

  v8f acc[2][2] = {};
  v16bf a0 = load_a_frag(Ar0, 1024, lane);
  v16bf a1 = load_a_frag(Ar1, 1024, lane);
  v16bf b0 = load_b_frag(Br0, 1024, lane);
  v16bf b1 = load_b_frag(Br1, 1024, lane);

  for (int k0 = 32; k0 < 1024; k0 += 32) {
    v16bf na0 = load_a_frag(Ar0 + k0, 1024, lane);
    v16bf na1 = load_a_frag(Ar1 + k0, 1024, lane);
    v16bf nb0 = load_b_frag(Br0 + k0, 1024, lane);
    v16bf nb1 = load_b_frag(Br1 + k0, 1024, lane);
    acc[0][0] = WMMA_BF16(a0, b0, acc[0][0]);
    acc[0][1] = WMMA_BF16(a0, b1, acc[0][1]);
    acc[1][0] = WMMA_BF16(a1, b0, acc[1][0]);
    acc[1][1] = WMMA_BF16(a1, b1, acc[1][1]);
    a0 = na0; a1 = na1; b0 = nb0; b1 = nb1;
  }
  acc[0][0] = WMMA_BF16(a0, b0, acc[0][0]);
  acc[0][1] = WMMA_BF16(a0, b1, acc[0][1]);
  acc[1][0] = WMMA_BF16(a1, b0, acc[1][0]);
  acc[1][1] = WMMA_BF16(a1, b1, acc[1][1]);

  const int roff = (lane & 16) ? 8 : 0;
#pragma unroll
  for (int mi = 0; mi < 2; ++mi) {
    int mbase = m0 + 16 * mi + roff;
#pragma unroll
    for (int ni = 0; ni < 2; ++ni) {
      int n = n0 + 16 * ni + (lane & 15);
#pragma unroll
      for (int i = 0; i < 8; ++i)
        Out[(size_t)(mbase + i) * 1024 + n] = acc[mi][ni][i];
    }
  }
}

// ------------------------------ launch -------------------------------------
extern "C" void kernel_launch(void* const* d_in, const int* in_sizes, int n_in,
                              void* d_out, int out_size, void* d_ws, size_t ws_size,
                              hipStream_t stream) {
  const float* x      = (const float*)d_in[0];   // [2,2048,1024]
  const float* w_qkv  = (const float*)d_in[1];   // [1024,3072]
  const float* w_proj = (const float*)d_in[2];   // [1024,1024]
  const float* rel    = (const float*)d_in[3];   // [32,16]

  char* ws = (char*)d_ws;
  size_t off = 0;
  auto alloc = [&](size_t bytes) -> void* {
    off = (off + 255) & ~(size_t)255;
    void* p = ws + off;
    off += bytes;
    return p;
  };

  const size_t M = 4096;               // B*T
  bf16_t* xb     = (bf16_t*)alloc(M * 1024 * 2);
  bf16_t* wqkvT  = (bf16_t*)alloc((size_t)3072 * 1024 * 2);
  bf16_t* wprojT = (bf16_t*)alloc((size_t)1024 * 1024 * 2);
  bf16_t* Qb     = (bf16_t*)alloc((size_t)2 * 16 * 2048 * 64 * 2);
  bf16_t* Kb     = (bf16_t*)alloc((size_t)2 * 16 * 2048 * 64 * 2);
  bf16_t* Vt     = (bf16_t*)alloc((size_t)2 * 16 * 64 * 2048 * 2);
  bf16_t* Yb     = (bf16_t*)alloc(M * 1024 * 2);
  float*  biasd  = (float*)alloc((size_t)16 * 2048 * 4);
  (void)ws_size; (void)in_sizes; (void)n_in; (void)out_size;

  // stage 0: conversions + bias table
  {
    int n = 2 * 2048 * 1024;
    k_cvt<<<(n + 255) / 256, 256, 0, stream>>>(x, xb, n);
    int nw = 1024 * 3072;
    k_cvt_T<<<(nw + 255) / 256, 256, 0, stream>>>(w_qkv, wqkvT, 1024, 3072);
    int np = 1024 * 1024;
    k_cvt_T<<<(np + 255) / 256, 256, 0, stream>>>(w_proj, wprojT, 1024, 1024);
    k_bias<<<8, 256, 0, stream>>>(rel, biasd);
  }
  // stage 1: QKV GEMM  (128 x 96 wave tiles, 8 waves/block)
  k_gemm_qkv<<<(128 * 96) / 8, 256, 0, stream>>>(xb, wqkvT, Qb, Kb, Vt);
  // stage 2: flash attention (4096 waves)
  k_attn<<<4096 / 8, 256, 0, stream>>>(Qb, Kb, Vt, biasd, Yb);
  // stage 3: output projection (128 x 32 wave tiles)
  k_gemm_proj<<<(128 * 32) / 8, 256, 0, stream>>>(Yb, wprojT, (float*)d_out);
}
